// CrfRnn_82179904242160
// MI455X (gfx1250) — matrix operand order repaired and checked by reading, not compiled, not run on hardware
//
#include <hip/hip_runtime.h>

#define HH 80
#define WW 80
#define NPIX (HH * WW)          // 6400
#define LBL 21
#define LPAD 32                 // labels padded to 2 WMMA M-tiles
#define NUM_ITER 5
#define ALPHA_ 160.0f
#define BETA_ 3.0f
#define GAMMA_ 3.0f
#define EPS_ 1e-8f
#define CHUNKS 8
#define JSPAN (NPIX / CHUNKS)   // 800 (25 K-steps of 32)

typedef __bf16 bf16_t;
typedef __attribute__((ext_vector_type(16))) __bf16 v16bf;
typedef __attribute__((ext_vector_type(8)))  __bf16 v8bf;
typedef __attribute__((ext_vector_type(8)))  float  v8f;

static __device__ __forceinline__ unsigned short f2bf(float f) {
    unsigned int u = __float_as_uint(f);
    u += 0x7FFFu + ((u >> 16) & 1u);       // round-to-nearest-even
    return (unsigned short)(u >> 16);
}

// ---------------------------------------------------------------------------
// Build K_sp / K_bi as bf16 (both together fit the 192MB L2), plus row sums.
// blockIdx.x = row j, blockIdx.y = kernel type (0 spatial, 1 bilateral)
// ---------------------------------------------------------------------------
__global__ void __launch_bounds__(256)
build_kernels(const float* __restrict__ image,
              unsigned short* __restrict__ Ksp,
              unsigned short* __restrict__ Kbi,
              float* __restrict__ rssp,
              float* __restrict__ rsbi) {
    const int j  = blockIdx.x;
    const int kt = blockIdx.y;
    const int t  = threadIdx.x;
    const float xj = (float)(j % WW), yj = (float)(j / WW);
    float sum = 0.f;
    if (kt == 0) {
        const float ig2 = 1.0f / (GAMMA_ * GAMMA_);
        for (int i = t; i < NPIX; i += 256) {
            float dx = (float)(i % WW) - xj;
            float dy = (float)(i / WW) - yj;
            float d2 = (dx * dx + dy * dy) * ig2;
            float v  = __expf(-0.5f * d2);
            Ksp[(size_t)j * NPIX + i] = f2bf(v);
            sum += v;
        }
    } else {
        const float ia2 = 1.0f / (ALPHA_ * ALPHA_);
        const float ib2 = 1.0f / (BETA_ * BETA_);
        const float rj = image[0 * NPIX + j];
        const float gj = image[1 * NPIX + j];
        const float bj = image[2 * NPIX + j];
        for (int i = t; i < NPIX; i += 256) {
            float dx = (float)(i % WW) - xj;
            float dy = (float)(i / WW) - yj;
            float dr = image[0 * NPIX + i] - rj;
            float dg = image[1 * NPIX + i] - gj;
            float db = image[2 * NPIX + i] - bj;
            float d2 = (dx * dx + dy * dy) * ia2 +
                       (dr * dr + dg * dg + db * db) * ib2;
            float v  = __expf(-0.5f * d2);
            Kbi[(size_t)j * NPIX + i] = f2bf(v);
            sum += v;
        }
    }
    __shared__ float red[256];
    red[t] = sum;
    __syncthreads();
    for (int s = 128; s > 0; s >>= 1) {
        if (t < s) red[t] += red[t + s];
        __syncthreads();
    }
    if (t == 0) (kt == 0 ? rssp : rsbi)[j] = red[0];
}

// ---------------------------------------------------------------------------
// Softmax over labels per pixel -> bf16 q, padded to 32 rows (rows 21..31 = 0)
// ---------------------------------------------------------------------------
__global__ void __launch_bounds__(256)
softmax_q(const float* __restrict__ cur, unsigned short* __restrict__ q) {
    const int i = blockIdx.x * blockDim.x + threadIdx.x;
    float v[LBL];
    float m = -1e30f;
    #pragma unroll
    for (int l = 0; l < LBL; ++l) { v[l] = cur[(size_t)l * NPIX + i]; m = fmaxf(m, v[l]); }
    float s = 0.f;
    #pragma unroll
    for (int l = 0; l < LBL; ++l) { v[l] = __expf(v[l] - m); s += v[l]; }
    const float inv = 1.0f / s;
    #pragma unroll
    for (int l = 0; l < LBL; ++l) q[(size_t)l * NPIX + i] = f2bf(v[l] * inv);
    #pragma unroll
    for (int l = LBL; l < LPAD; ++l) q[(size_t)l * NPIX + i] = 0;
}

// ---------------------------------------------------------------------------
// WMMA GEMM: out[m, i] = sum_j q[m, j] * K[j, i] for both kernels.
// Block = 8 waves; wave w handles K-chunk w of TWO adjacent i-tiles
// (32 output columns), reusing the A (q) operand across both tiles to halve
// redundant q traffic from L2. B uses K's symmetry: column i of K == row i.
// ---------------------------------------------------------------------------
__global__ void __launch_bounds__(256)
gemm_qk(const unsigned short* __restrict__ qraw,
        const unsigned short* __restrict__ Kspr,
        const unsigned short* __restrict__ Kbir,
        float* __restrict__ partial) {
    const bf16_t* q   = (const bf16_t*)qraw;
    const bf16_t* Ksp = (const bf16_t*)Kspr;
    const bf16_t* Kbi = (const bf16_t*)Kbir;

    const int i0    = blockIdx.x * 32;     // two 16-col tiles per wave
    const int chunk = threadIdx.x >> 5;
    const int lane  = threadIdx.x & 31;
    const int mlo   = lane & 15;           // A: row within half, B: column
    const int hi    = lane >> 4;           // lane half
    const int hi8   = hi * 8;

    // accumulators: [tile 0/1][M-block 0/1] x [sp, bi]
    v8f s0a = {}, s0b = {}, s1a = {}, s1b = {};
    v8f b0a = {}, b0b = {}, b1a = {}, b1b = {};

    const bf16_t* qr0 = q + (size_t)mlo * NPIX;          // M rows 0..15
    const bf16_t* qr1 = q + (size_t)(mlo + 16) * NPIX;   // M rows 16..31
    const bf16_t* bsp0 = Ksp + (size_t)(i0 + mlo)      * NPIX + hi * 16;
    const bf16_t* bsp1 = Ksp + (size_t)(i0 + 16 + mlo) * NPIX + hi * 16;
    const bf16_t* bbi0 = Kbi + (size_t)(i0 + mlo)      * NPIX + hi * 16;
    const bf16_t* bbi1 = Kbi + (size_t)(i0 + 16 + mlo) * NPIX + hi * 16;

    const int jbase = chunk * JSPAN;
    for (int js = 0; js < JSPAN; js += 32) {
        const int jj = jbase + js;
        // A (16x32 bf16): per lane two 16B chunks at K-offsets hi*16, hi*16+8
        v8bf a0lo = *(const v8bf*)(qr0 + jj + hi * 16);
        v8bf a0hi = *(const v8bf*)(qr0 + jj + hi * 16 + 8);
        v8bf a1lo = *(const v8bf*)(qr1 + jj + hi * 16);
        v8bf a1hi = *(const v8bf*)(qr1 + jj + hi * 16 + 8);
        v16bf a0 = __builtin_shufflevector(a0lo, a0hi,
                     0,1,2,3,4,5,6,7,8,9,10,11,12,13,14,15);
        v16bf a1 = __builtin_shufflevector(a1lo, a1hi,
                     0,1,2,3,4,5,6,7,8,9,10,11,12,13,14,15);
        // B (32x16 bf16): lane holds column, K rows jj+hi*16..+15 contiguous
        v16bf bs0 = *(const v16bf*)(bsp0 + jj);
        v16bf bs1 = *(const v16bf*)(bsp1 + jj);
        v16bf bb0 = *(const v16bf*)(bbi0 + jj);
        v16bf bb1 = *(const v16bf*)(bbi1 + jj);

        s0a = __builtin_amdgcn_wmma_f32_16x16x32_bf16(false, a0, false, bs0,
                                                      (short)0, s0a, false, false);
        s0b = __builtin_amdgcn_wmma_f32_16x16x32_bf16(false, a1, false, bs0,
                                                      (short)0, s0b, false, false);
        s1a = __builtin_amdgcn_wmma_f32_16x16x32_bf16(false, a0, false, bs1,
                                                      (short)0, s1a, false, false);
        s1b = __builtin_amdgcn_wmma_f32_16x16x32_bf16(false, a1, false, bs1,
                                                      (short)0, s1b, false, false);
        b0a = __builtin_amdgcn_wmma_f32_16x16x32_bf16(false, a0, false, bb0,
                                                      (short)0, b0a, false, false);
        b0b = __builtin_amdgcn_wmma_f32_16x16x32_bf16(false, a1, false, bb0,
                                                      (short)0, b0b, false, false);
        b1a = __builtin_amdgcn_wmma_f32_16x16x32_bf16(false, a0, false, bb1,
                                                      (short)0, b1a, false, false);
        b1b = __builtin_amdgcn_wmma_f32_16x16x32_bf16(false, a1, false, bb1,
                                                      (short)0, b1b, false, false);
    }

    // C/D layout: VGPR r holds M=r (lanes 0-15) / M=r+8 (lanes 16-31), N=lane%16
    const int col0  = i0 + mlo;
    const int col1  = i0 + 16 + mlo;
    const int rbase = hi ? 8 : 0;
    float* psp = partial + (size_t)(0 * CHUNKS + chunk) * LPAD * NPIX;
    float* pbi = partial + (size_t)(1 * CHUNKS + chunk) * LPAD * NPIX;
    #pragma unroll
    for (int r = 0; r < 8; ++r) {
        psp[(size_t)(rbase + r)      * NPIX + col0] = s0a[r];
        psp[(size_t)(rbase + r + 16) * NPIX + col0] = s0b[r];
        psp[(size_t)(rbase + r)      * NPIX + col1] = s1a[r];
        psp[(size_t)(rbase + r + 16) * NPIX + col1] = s1b[r];
        pbi[(size_t)(rbase + r)      * NPIX + col0] = b0a[r];
        pbi[(size_t)(rbase + r + 16) * NPIX + col0] = b0b[r];
        pbi[(size_t)(rbase + r)      * NPIX + col1] = b1a[r];
        pbi[(size_t)(rbase + r + 16) * NPIX + col1] = b1b[r];
    }
}

// ---------------------------------------------------------------------------
// Reduce chunks, apply norms, label-weight matrices, compatibility, + unary.
// One thread per pixel.
// ---------------------------------------------------------------------------
__global__ void __launch_bounds__(256)
combine(const float* __restrict__ partial,
        const float* __restrict__ rssp, const float* __restrict__ rsbi,
        const float* __restrict__ unary,
        const float* __restrict__ Wsp, const float* __restrict__ Wbi,
        const float* __restrict__ Cm,
        float* __restrict__ cur) {
    const int i = blockIdx.x * blockDim.x + threadIdx.x;
    const float nsp = 1.0f / (rssp[i] + EPS_);
    const float nbi = 1.0f / (rsbi[i] + EPS_);
    float ssp[LBL], sbi[LBL];
    #pragma unroll
    for (int l = 0; l < LBL; ++l) {
        float a = 0.f, b = 0.f;
        #pragma unroll
        for (int c = 0; c < CHUNKS; ++c) {
            a += partial[((size_t)(0 * CHUNKS + c) * LPAD + l) * NPIX + i];
            b += partial[((size_t)(1 * CHUNKS + c) * LPAD + l) * NPIX + i];
        }
        ssp[l] = a * nsp;
        sbi[l] = b * nbi;
    }
    float tt[LBL];
    #pragma unroll
    for (int l = 0; l < LBL; ++l) {
        float acc = 0.f;
        #pragma unroll
        for (int lp = 0; lp < LBL; ++lp)
            acc += Wsp[l * LBL + lp] * ssp[lp] + Wbi[l * LBL + lp] * sbi[lp];
        tt[l] = acc;
    }
    #pragma unroll
    for (int l = 0; l < LBL; ++l) {
        float acc = unary[(size_t)l * NPIX + i];
        #pragma unroll
        for (int lp = 0; lp < LBL; ++lp)
            acc += Cm[l * LBL + lp] * tt[lp];
        cur[(size_t)l * NPIX + i] = acc;
    }
}

// ---------------------------------------------------------------------------
extern "C" void kernel_launch(void* const* d_in, const int* in_sizes, int n_in,
                              void* d_out, int out_size, void* d_ws, size_t ws_size,
                              hipStream_t stream) {
    const float* image  = (const float*)d_in[0];
    const float* logits = (const float*)d_in[1];
    const float* Wsp    = (const float*)d_in[2];
    const float* Wbi    = (const float*)d_in[3];
    const float* Cm     = (const float*)d_in[4];
    float* cur = (float*)d_out;   // cur lives in d_out; final cur IS the output

    char* w = (char*)d_ws;
    const size_t NN = (size_t)NPIX * NPIX;
    unsigned short* Ksp = (unsigned short*)w; w += NN * 2;                  // 82 MB
    unsigned short* Kbi = (unsigned short*)w; w += NN * 2;                  // 82 MB
    float* rssp = (float*)w;          w += (size_t)NPIX * 4;
    float* rsbi = (float*)w;          w += (size_t)NPIX * 4;
    unsigned short* q = (unsigned short*)w; w += (size_t)LPAD * NPIX * 2;   // 0.4 MB
    float* partial = (float*)w;       // 2*CHUNKS*LPAD*NPIX*4 = 13.1 MB

    build_kernels<<<dim3(NPIX, 2), 256, 0, stream>>>(image, Ksp, Kbi, rssp, rsbi);
    for (int it = 0; it < NUM_ITER; ++it) {
        const float* src = (it == 0) ? logits : cur;
        softmax_q<<<NPIX / 256, 256, 0, stream>>>(src, q);
        gemm_qk<<<NPIX / 32, 256, 0, stream>>>(q, Ksp, Kbi, partial);
        combine<<<NPIX / 256, 256, 0, stream>>>(partial, rssp, rsbi, logits,
                                                Wsp, Wbi, Cm, cur);
    }
}
